// EdgeUpdater_4372276707996
// MI455X (gfx1250) — matrix operand order repaired and checked by reading, not compiled, not run on hardware
//
#include <hip/hip_runtime.h>

// ---- CDNA5 vector types ----
typedef _Float16 v4h  __attribute__((ext_vector_type(4)));
typedef _Float16 v8h  __attribute__((ext_vector_type(8)));
typedef _Float16 v16h __attribute__((ext_vector_type(16)));
typedef float    v4f  __attribute__((ext_vector_type(4)));
typedef float    v8f  __attribute__((ext_vector_type(8)));
typedef unsigned int u32x4 __attribute__((ext_vector_type(4)));
typedef int          i32x4 __attribute__((ext_vector_type(4)));
typedef int          i32x8 __attribute__((ext_vector_type(8)));

#define HID       128
#define CONCAT    384
#define TILE_M    128
#define SA_STRIDE 392                         // 384+8 halves: bank-conflict-free rows
#define SW_STRIDE 136                         // 128+8 halves
#define PANEL_HALVES (HID * SW_STRIDE)        // 17408 halves
#define PANEL_BYTES  (PANEL_HALVES * 2)       // 34816 B per weight panel
#define N_PANELS  6                           // W0 = 3 panels, W1..W3 = 1 each

#define USE_TDM (__has_builtin(__builtin_amdgcn_tensor_load_to_lds) && \
                 __has_builtin(__builtin_amdgcn_s_wait_tensorcnt))

#if USE_TDM
#if defined(__clang_major__) && (__clang_major__ >= 23)
#define TDM_CALL(g0, g1, g2, g3) \
    do { i32x8 g4z = {}; __builtin_amdgcn_tensor_load_to_lds(g0, g1, g2, g3, g4z, 0); } while (0)
#else
#define TDM_CALL(g0, g1, g2, g3) \
    __builtin_amdgcn_tensor_load_to_lds(g0, g1, g2, g3, 0)
#endif

// 1-D TDM copy: PANEL_BYTES from global (d_ws, pre-converted f16) into LDS.
// D# per CDNA5 ISA ch.8: group0 = {count=1 | lds_addr | global_addr | type=2},
// group1 = {data_size=8B, tensor_dim0 = tile_dim0 = bytes/8, dims1..2 = 1}.
__device__ __forceinline__ void tdm_load_panel(void* ldsDst, const void* gsrc)
{
    unsigned           lds = (unsigned)(unsigned long long)ldsDst;   // addr[31:0] = LDS offset
    unsigned long long ga  = (unsigned long long)gsrc;
    const int n8 = PANEL_BYTES >> 3;                                 // 8-byte elements
    u32x4 g0;
    g0[0] = 1u;                                   // count=1 (valid, user mode)
    g0[1] = lds;                                  // lds_addr
    g0[2] = (unsigned)ga;                         // global_addr[31:0]
    g0[3] = (unsigned)(ga >> 32) | (2u << 30);    // global_addr[56:32] | type=2 (image)
    i32x8 g1;
    g1[0] = 3 << 16;                              // data_size = 8B; mask/flags = 0
    g1[1] = (n8 & 0xFFFF) << 16;                  // tensor_dim0[15:0]  (bits 63:48)
    g1[2] = (n8 >> 16) | (1 << 16);               // tensor_dim0[31:16] | tensor_dim1 = 1
    g1[3] = (n8 & 0xFFFF) << 16;                  // tile_dim0 = n8 (bits 127:112)
    g1[4] = 1;                                    // tile_dim1 = 1, tile_dim2 = 0
    g1[5] = n8;                                   // tensor_dim0_stride[31:0]
    g1[6] = 0;
    g1[7] = 0;
    i32x4 gz = {};
    TDM_CALL(g0, g1, gz, gz);
}
#endif

// ---------- prep: convert W0..W3 (fp32, [K][128] row-major) into d_ws as
// f16, transposed (ws[n][k]) with SW_STRIDE padding, one panel per 128 K-rows.
__global__ __launch_bounds__(256)
void prep_weights_kernel(const float* __restrict__ W0, const float* __restrict__ W1,
                         const float* __restrict__ W2, const float* __restrict__ W3,
                         _Float16* __restrict__ ws)
{
    int i = blockIdx.x * 256 + threadIdx.x;
    if (i >= N_PANELS * HID * HID) return;
    int panel = i >> 14;            // / 16384
    int r     = i & 16383;
    int k     = r >> 7;
    int n     = r & 127;
    const float* W;
    int krow;
    if (panel < 3)      { W = W0; krow = panel * HID + k; }
    else if (panel == 3){ W = W1; krow = k; }
    else if (panel == 4){ W = W2; krow = k; }
    else                { W = W3; krow = k; }
    ws[(size_t)panel * PANEL_HALVES + n * SW_STRIDE + k] =
        (_Float16)W[(size_t)krow * HID + n];
}

// ---------- main fused kernel --------------------------------------------
__global__ __launch_bounds__(256)
void edge_mlp_kernel(const float* __restrict__ x,
                     const float* __restrict__ edge,
                     const long long* __restrict__ eidx,
                     const float* __restrict__ b0, const float* __restrict__ b1,
                     const float* __restrict__ b2, const float* __restrict__ b3,
                     const _Float16* __restrict__ wsW,
                     float* __restrict__ out, int E)
{
    __shared__ __align__(16) _Float16 sA[TILE_M * SA_STRIDE]; // activations, f16
    __shared__ __align__(16) _Float16 sW[PANEL_HALVES];       // weight panel (f16, transposed)
    __shared__ float sBias[HID];
    __shared__ int   sSrc[TILE_M];
    __shared__ int   sDst[TILE_M];

    const int       tid   = threadIdx.x;
    const long long eBase = (long long)blockIdx.x * TILE_M;

    // ---- stage gather indices (edge_index: int64, [2, E]) ----
    if (tid < TILE_M) {
        long long e = eBase + tid; if (e >= E) e = E - 1;
        sSrc[tid] = (int)eidx[e];
    } else {
        int r = tid - TILE_M;
        long long e = eBase + r; if (e >= E) e = E - 1;
        sDst[r] = (int)eidx[(long long)E + e];
    }
    __syncthreads();

    // ---- gather + concat -> sA as f16:  [x[src] | x[dst] | edge] ----
    for (int c = tid; c < TILE_M * (CONCAT / 4); c += 256) {
        int row = c / (CONCAT / 4);
        int col = (c - row * (CONCAT / 4)) * 4;
        const float* p;
        if (col < HID) {
            p = x + (size_t)sSrc[row] * HID + col;
        } else if (col < 2 * HID) {
            p = x + (size_t)sDst[row] * HID + (col - HID);
        } else {
            long long e = eBase + row; if (e >= E) e = E - 1;
            p = edge + (size_t)e * HID + (col - 2 * HID);
        }
        v4f f = *(const v4f*)p;
        *(v4h*)(&sA[row * SA_STRIDE + col]) = __builtin_convertvector(f, v4h);
    }

    const int lane = tid & 31;
    const int wave = tid >> 5;
    const int rlo  = lane & 15;   // A row / B,C col within 16-tile
    const int hsel = lane >> 4;   // half-wave select
    const int m0   = wave * 16;   // this wave's 16-row output strip
    const int mA   = m0 + rlo;

    // ---- one layer:  sA[:,0..K) @ W[K,128] (+bias, optional relu/residual) ----
    auto run_layer = [&](int K, int panel0, const float* __restrict__ b,
                         bool relu, bool last) {
        v8f acc[8];
        const v8f vzero = {};
        #pragma unroll
        for (int t = 0; t < 8; ++t) acc[t] = vzero;

        for (int p = 0; p < K; p += HID) {       // 128-wide K panels
            __syncthreads();                     // previous sW fully consumed
            const _Float16* wsPanel = wsW + (size_t)(panel0 + (p >> 7)) * PANEL_HALVES;
#if USE_TDM
            if (wave == 0) {
                tdm_load_panel(sW, wsPanel);     // async DMA, TENSORcnt++
            }
#else
            {   // fallback: plain b128 LDS fill (no conversion needed, ws is f16)
                const v8h* s = (const v8h*)wsPanel;
                v8h*       d = (v8h*)sW;
                for (int i = tid; i < PANEL_HALVES / 8; i += 256) d[i] = s[i];
            }
#endif
            if (p == 0 && tid < HID) sBias[tid] = b[tid];
#if USE_TDM
            if (wave == 0) __builtin_amdgcn_s_wait_tensorcnt(0);
#endif
            __syncthreads();

            #pragma unroll
            for (int kt = 0; kt < HID; kt += 32) {
                // A fragment (16x32 f16): row mA, K = p+kt+hsel*8+[0..8) and +16
                const _Float16* pa = &sA[mA * SA_STRIDE + p + kt + hsel * 8];
                v8h a0 = *(const v8h*)pa;
                v8h a1 = *(const v8h*)(pa + 16);
                v16h Af = __builtin_shufflevector(a0, a1,
                    0,1,2,3,4,5,6,7,8,9,10,11,12,13,14,15);
                // B fragments, software-pipelined one tile ahead
                const _Float16* pb0 = &sW[rlo * SW_STRIDE + kt + hsel * 8];
                v8h c0 = *(const v8h*)pb0;
                v8h c1 = *(const v8h*)(pb0 + 16);
                #pragma unroll
                for (int tc = 0; tc < 8; ++tc) {
                    v8h n0 = c0, n1 = c1;
                    if (tc < 7) {
                        const _Float16* pb = pb0 + (tc + 1) * 16 * SW_STRIDE;
                        n0 = *(const v8h*)pb;
                        n1 = *(const v8h*)(pb + 16);
                    }
                    v16h Bf = __builtin_shufflevector(c0, c1,
                        0,1,2,3,4,5,6,7,8,9,10,11,12,13,14,15);
                    acc[tc] = __builtin_amdgcn_wmma_f32_16x16x32_f16(
                        false, Af, false, Bf, (short)0, acc[tc], false, false);
                    c0 = n0; c1 = n1;
                }
            }
        }
        __syncthreads();   // all sA reads done before overwrite

        if (!last) {
            #pragma unroll
            for (int tc = 0; tc < 8; ++tc) {
                int n = tc * 16 + rlo;
                float bn = sBias[n];
                #pragma unroll
                for (int i = 0; i < 8; ++i) {
                    float v = acc[tc][i] + bn;            // C/D row = m0 + hsel*8 + i
                    if (relu) v = v > 0.f ? v : 0.f;
                    sA[(m0 + hsel * 8 + i) * SA_STRIDE + n] = (_Float16)v;
                }
            }
            __syncthreads();
        } else {
            // final layer: residual  out = edge + h  (fp32, guarded tail)
            #pragma unroll
            for (int i = 0; i < 8; ++i) {
                long long e = eBase + m0 + hsel * 8 + i;
                if (e < E) {
                    #pragma unroll
                    for (int tc = 0; tc < 8; ++tc) {
                        int n = tc * 16 + rlo;
                        size_t off = (size_t)e * HID + n;
                        out[off] = edge[off] + (acc[tc][i] + sBias[n]);
                    }
                }
            }
        }
    };

    run_layer(CONCAT, 0, b0, /*relu=*/true,  /*last=*/false);  // panels 0..2
    run_layer(HID,    3, b1, /*relu=*/true,  /*last=*/false);
    run_layer(HID,    4, b2, /*relu=*/true,  /*last=*/false);
    run_layer(HID,    5, b3, /*relu=*/false, /*last=*/true);
}

extern "C" void kernel_launch(void* const* d_in, const int* in_sizes, int n_in,
                              void* d_out, int out_size, void* d_ws, size_t ws_size,
                              hipStream_t stream)
{
    (void)n_in; (void)ws_size; (void)out_size;
    const float*     x    = (const float*)d_in[0];
    const float*     edge = (const float*)d_in[1];
    const long long* eidx = (const long long*)d_in[2];   // int64 per reference
    const float*     W0   = (const float*)d_in[3];
    const float*     b0   = (const float*)d_in[4];
    const float*     W1   = (const float*)d_in[5];
    const float*     b1   = (const float*)d_in[6];
    const float*     W2   = (const float*)d_in[7];
    const float*     b2   = (const float*)d_in[8];
    const float*     W3   = (const float*)d_in[9];
    const float*     b3   = (const float*)d_in[10];

    _Float16* wsW = (_Float16*)d_ws;   // needs N_PANELS*PANEL_BYTES = 208,896 B scratch

    int E    = in_sizes[1] / HID;      // edge is [E, 128]
    int grid = (E + TILE_M - 1) / TILE_M;

    hipLaunchKernelGGL(prep_weights_kernel,
                       dim3((N_PANELS * HID * HID + 255) / 256), dim3(256), 0, stream,
                       W0, W1, W2, W3, wsW);
    hipLaunchKernelGGL(edge_mlp_kernel, dim3(grid), dim3(256), 0, stream,
                       x, edge, eidx, b0, b1, b2, b3, wsW,
                       (float*)d_out, E);
}
